// HybridBackbone_81990925681360
// MI455X (gfx1250) — matrix operand order repaired and checked by reading, not compiled, not run on hardware
//
#include <hip/hip_runtime.h>
#include <math.h>

// ---------------------------------------------------------------------------
// HybridBackbone for MI455X (gfx1250): bf16 WMMA GEMM core + VALU glue kernels
// ---------------------------------------------------------------------------

typedef __bf16 bf16_t;
typedef __attribute__((ext_vector_type(16))) __bf16 v16bf;
typedef __attribute__((ext_vector_type(8)))  __bf16 v8bf;
typedef __attribute__((ext_vector_type(8)))  float  v8f;

#define BB 32
#define NN 192
#define HH 256
#define KNBR 32
#define NNODE (BB * NN)          // 6144
#define NEDGE (NNODE * KNBR)     // 196608
#define ECHUNK 8192              // edges per chunk (=> 256 nodes per chunk)

__device__ __forceinline__ float silu_f(float x) { return x / (1.0f + __expf(-x)); }
__device__ __forceinline__ float gelu_f(float x) { return 0.5f * x * (1.0f + erff(x * 0.70710678118654752f)); }

// ===========================================================================
// WMMA GEMM: C[M,N] = act(A[M,K] @ W[K,N] + bias [+ res])
//   Requires: M % 128 == 0, N % 128 == 0, K % 32 == 0 (all call sites comply).
// MODE 0: plain A[M,K].  MODE 1: gathered GNN-m1 input (K=768):
//   cols 0-255 -> x[self], 256-511 -> x[neighbor via idx], 512-767 -> edge feat.
// ACT: 0 none, 1 silu, 2 exact gelu.  RES: compile-time residual add.
// 256 threads = 8 waves; tile 128x128; wave owns 32x64 (2x4 WMMA accums).
// Double-buffered LDS; A staged row-major, B staged TRANSPOSED so both
// fragment types are two 16B chunks (k 8h.., 16+8h..) -> ds_load_b128 pairs.
// Row stride 40 bf16 (80B) keeps every fragment chunk 16B-aligned.
// ===========================================================================
#define BM 128
#define BN 128
#define BK 32
#define LDSS 40   // LDS row stride in bf16 elements (80 bytes)

union FragU { v16bf v; v8bf h[2]; };

__device__ __forceinline__ v16bf load_frag(const bf16_t* base, int hi_off) {
  FragU u;
  u.h[0] = *(const v8bf*)(base);
  u.h[1] = *(const v8bf*)(base + hi_off);
  return u.v;
}

template <int MODE, int ACT, bool RES>
__global__ __launch_bounds__(256) void gemm_wmma(
    const float* __restrict__ A, const float* __restrict__ W,
    const float* __restrict__ bias, const float* __restrict__ res,
    float* __restrict__ C, int M, int N, int K, int ldC, int coff,
    const float* __restrict__ gx, const float* __restrict__ gef,
    const int* __restrict__ gidx, int edge0)
{
  __shared__ bf16_t As[2][BM][LDSS];
  __shared__ bf16_t Bs[2][BN][LDSS];   // transposed: Bs[col][k]

  const int tid  = threadIdx.x;
  const int lane = tid & 31;
  const int wid  = tid >> 5;
  const int wm   = wid & 3;        // 4 row-waves (32 rows each)
  const int wn   = wid >> 2;       // 2 col-waves (64 cols each)
  const int m0   = blockIdx.y * BM;
  const int n0   = blockIdx.x * BN;
  const int hk   = lane >> 4;      // lane half
  const int l16  = lane & 15;

  v8f acc[2][4] = {};
  float a_reg[16], b_reg[16];

  const int nk = K / BK;

  // ---- register-stage loaders (no bounds checks: exact tiling) ----
  auto stage_load = [&](int k0) {
    if (MODE == 0) {
#pragma unroll
      for (int i = 0; i < 16; ++i) {
        int lin = i * 256 + tid;
        int r = lin >> 5, c = lin & 31;
        a_reg[i] = A[(m0 + r) * K + (k0 + c)];
      }
    } else {
      int seg = k0 >> 8;          // scalar per k-tile: concat segment
      int kc  = k0 & 255;
#pragma unroll
      for (int i = 0; i < 16; ++i) {
        int lin = i * 256 + tid;
        int r = lin >> 5, c = lin & 31;
        int row = m0 + r;
        if (seg == 0) {
          int node = (edge0 + row) >> 5;
          a_reg[i] = gx[node * 256 + kc + c];
        } else if (seg == 1) {
          int e = edge0 + row;
          int node = e >> 5;
          int bb = node / NN;
          int nl = gidx[e];
          a_reg[i] = gx[(bb * NN + nl) * 256 + kc + c];
        } else {
          a_reg[i] = gef[row * 256 + kc + c];
        }
      }
    }
#pragma unroll
    for (int i = 0; i < 16; ++i) {
      int lin = i * 256 + tid;
      int r = lin >> 7, c = lin & 127;
      b_reg[i] = W[(k0 + r) * N + (n0 + c)];
    }
  };
  auto stage_store = [&](int buf) {
#pragma unroll
    for (int i = 0; i < 16; ++i) {
      int lin = i * 256 + tid;
      As[buf][lin >> 5][lin & 31] = (bf16_t)a_reg[i];
    }
#pragma unroll
    for (int i = 0; i < 16; ++i) {
      int lin = i * 256 + tid;
      Bs[buf][lin & 127][lin >> 7] = (bf16_t)b_reg[i];   // transpose on store
    }
  };

  // ---- pipelined main loop ----
  stage_load(0);
  stage_store(0);
  int cur = 0;
  for (int step = 0; step < nk; ++step) {
    __syncthreads();
    if (step + 1 < nk) stage_load((step + 1) * BK);              // HBM -> regs
    if (step + 2 < nk)
      __builtin_prefetch(&W[((step + 2) * BK) * N + n0 + (tid & 127)], 0, 1);

    // fragments: two 16B chunks (k 8h+s, 16+8h+s) -> b128 LDS loads
    v16bf afr[2], bfr[4];
#pragma unroll
    for (int r = 0; r < 2; ++r)
      afr[r] = load_frag(&As[cur][wm * 32 + r * 16 + l16][hk * 8], 16);
#pragma unroll
    for (int c = 0; c < 4; ++c)
      bfr[c] = load_frag(&Bs[cur][wn * 64 + c * 16 + l16][hk * 8], 16);
#pragma unroll
    for (int r = 0; r < 2; ++r)
#pragma unroll
      for (int c = 0; c < 4; ++c)
        acc[r][c] = __builtin_amdgcn_wmma_f32_16x16x32_bf16(
            false, afr[r], false, bfr[c], (short)0, acc[r][c], false, false);

    if (step + 1 < nk) stage_store(cur ^ 1);                     // regs -> LDS
    cur ^= 1;
  }

  // ---- epilogue (branch-free: exact tiling, templated residual) ----
#pragma unroll
  for (int r = 0; r < 2; ++r) {
#pragma unroll
    for (int c = 0; c < 4; ++c) {
      int col = n0 + wn * 64 + c * 16 + l16;
      float bv = bias[col];
#pragma unroll
      for (int v = 0; v < 8; ++v) {
        int row = m0 + wm * 32 + r * 16 + hk * 8 + v;
        float val = acc[r][c][v] + bv;
        if (RES) val += res[row * N + col];
        if (ACT == 1) val = silu_f(val);
        else if (ACT == 2) val = gelu_f(val);
        C[row * ldC + coff + col] = val;
      }
    }
  }
}

// ===========================================================================
// Small glue kernels
// ===========================================================================
__global__ void zero_flag_kernel(int* flag) { *flag = 0; }

__global__ void theta_flag_kernel(const float* __restrict__ th, int* flag, int ntot) {
  int i = blockIdx.x * 256 + threadIdx.x;
  if (i < ntot) {
    float v = th[i];
    if (v < 0.0f || v > 1.0f) atomicOr(flag, 1);
  }
}

__global__ void theta_softmax_kernel(const float* __restrict__ th, float* __restrict__ out,
                                     const int* __restrict__ flag, int rows) {
  int r = blockIdx.x * 256 + threadIdx.x;
  if (r >= rows) return;
  const float* x = th + r * 16;
  float* y = out + r * 16;
  if (*flag) {
    float mx = -3.0e38f;
#pragma unroll
    for (int i = 0; i < 16; ++i) mx = fmaxf(mx, x[i]);
    float e[16], s = 0.0f;
#pragma unroll
    for (int i = 0; i < 16; ++i) { e[i] = __expf(x[i] - mx); s += e[i]; }
    float inv = 1.0f / s;
#pragma unroll
    for (int i = 0; i < 16; ++i) y[i] = e[i] * inv;
  } else {
#pragma unroll
    for (int i = 0; i < 16; ++i) y[i] = x[i];
  }
}

// sinusoidal time embedding + 2-layer MLP + projection, one workgroup (tiny)
__global__ void time_embed_kernel(const float* __restrict__ t,
                                  const float* __restrict__ W1, const float* __restrict__ b1,
                                  const float* __restrict__ W2, const float* __restrict__ b2,
                                  const float* __restrict__ Wp, const float* __restrict__ bp,
                                  float* __restrict__ temb) {
  __shared__ float emb[32 * 128];
  __shared__ float hid[32 * 128];
  int tid = threadIdx.x;
  for (int i = tid; i < 32 * 128; i += 256) {
    int b = i >> 7, c = i & 127;
    float val;
    if (c < 64) val = __sinf(t[b] * powf(10000.0f, -(float)c / 64.0f));
    else        val = __cosf(t[b] * powf(10000.0f, -(float)(c - 64) / 64.0f));
    emb[i] = val;
  }
  __syncthreads();
  for (int i = tid; i < 32 * 128; i += 256) {
    int b = i >> 7, c = i & 127;
    float a = b1[c];
    for (int k = 0; k < 128; ++k) a += emb[b * 128 + k] * W1[k * 128 + c];
    hid[i] = silu_f(a);
  }
  __syncthreads();
  for (int i = tid; i < 32 * 128; i += 256) {
    int b = i >> 7, c = i & 127;
    float a = b2[c];
    for (int k = 0; k < 128; ++k) a += hid[b * 128 + k] * W2[k * 128 + c];
    emb[i] = a;
  }
  __syncthreads();
  for (int i = tid; i < 32 * 256; i += 256) {
    int b = i >> 8, c = i & 255;
    float a = bp[c];
    for (int k = 0; k < 128; ++k) a += emb[b * 128 + k] * Wp[k * 256 + c];
    temb[i] = a;
  }
}

// base = theta_s @ Wnode + bnode + temb[batch]
__global__ void node_base_kernel(const float* __restrict__ thetas, const float* __restrict__ Wn,
                                 const float* __restrict__ bn, const float* __restrict__ temb,
                                 const int* __restrict__ batch, float* __restrict__ out) {
  __shared__ float th[16];
  int i = blockIdx.x, j = threadIdx.x;
  if (j < 16) th[j] = thetas[i * 16 + j];
  __syncthreads();
  float a = bn[j];
#pragma unroll
  for (int k = 0; k < 16; ++k) a += th[k] * Wn[k * 256 + j];
  a += temb[batch[i] * 256 + j];
  out[i * 256 + j] = a;
}

// h = silu(pos @ W1 + b1)   (K=3)
__global__ void pos_hidden_kernel(const float* __restrict__ pos, const float* __restrict__ W1,
                                  const float* __restrict__ b1, float* __restrict__ h) {
  int i = blockIdx.x, j = threadIdx.x;
  const float* p = pos + i * 3;
  float a = b1[j] + p[0] * W1[j] + p[1] * W1[256 + j] + p[2] * W1[512 + j];
  h[i * 256 + j] = silu_f(a);
}

// KNN: per (b, n) row, K=32 smallest d^2 (self +1e9), tie -> lower index
__global__ void knn_kernel(const float* __restrict__ pos, int* __restrict__ idxb,
                           float* __restrict__ maskb) {
  __shared__ float ps[NN * 3];
  __shared__ float d2s[256];
  __shared__ float rv[256];
  __shared__ int   ri[256];
  int n = blockIdx.x, b = blockIdx.y, tid = threadIdx.x;
  for (int i = tid; i < NN * 3; i += 256) ps[i] = pos[(b * NN) * 3 + i];
  __syncthreads();
  float px = ps[n * 3], py = ps[n * 3 + 1], pz = ps[n * 3 + 2];
  if (tid < NN) {
    float dx = ps[tid * 3] - px, dy = ps[tid * 3 + 1] - py, dz = ps[tid * 3 + 2] - pz;
    d2s[tid] = dx * dx + dy * dy + dz * dz + (tid == n ? 1.0e9f : 0.0f);
  } else {
    d2s[tid] = 3.0e38f;
  }
  __syncthreads();
  for (int it = 0; it < KNBR; ++it) {
    rv[tid] = d2s[tid];
    ri[tid] = tid;
    __syncthreads();
    for (int off = 128; off >= 1; off >>= 1) {
      if (tid < off) {
        float v2 = rv[tid + off]; int i2 = ri[tid + off];
        if (v2 < rv[tid] || (v2 == rv[tid] && i2 < ri[tid])) { rv[tid] = v2; ri[tid] = i2; }
      }
      __syncthreads();
    }
    if (tid == 0) {
      int im = ri[0]; float vm = rv[0];
      idxb[(b * NN + n) * KNBR + it]  = im;
      maskb[(b * NN + n) * KNBR + it] = (vm <= 25.0f) ? 1.0f : 0.0f;  // CUTOFF^2
      d2s[im] = 3.0e38f;
    }
    __syncthreads();
  }
}

// per-edge geometry: {dist, dir_x, dir_y, dir_z}
__global__ void edge_geom_kernel(const float* __restrict__ pos, const int* __restrict__ idxb,
                                 float* __restrict__ geom, int edge0, int nE) {
  int el = blockIdx.x * 256 + threadIdx.x;
  if (el >= nE) return;
  int e = edge0 + el;
  int node = e >> 5;
  int b = node / NN, n = node - b * NN;
  int nl = idxb[e];
  const float* pi = pos + (b * NN + n) * 3;
  const float* pj = pos + (b * NN + nl) * 3;
  float dx = pj[0] - pi[0], dy = pj[1] - pi[1], dz = pj[2] - pi[2];
  float dist = sqrtf(dx * dx + dy * dy + dz * dz);
  float inv = 1.0f / (dist + 1e-8f);
  geom[el * 4 + 0] = dist;
  geom[el * 4 + 1] = dx * inv;
  geom[el * 4 + 2] = dy * inv;
  geom[el * 4 + 3] = dz * inv;
}

__global__ void dist_hidden_kernel(const float* __restrict__ geom, const float* __restrict__ W,
                                   const float* __restrict__ b, float* __restrict__ h) {
  int e = blockIdx.x, j = threadIdx.x;  // 128 threads
  h[e * 128 + j] = silu_f(geom[e * 4] * W[j] + b[j]);
}

__global__ void dir_hidden_kernel(const float* __restrict__ geom, const float* __restrict__ W,
                                  const float* __restrict__ b, float* __restrict__ h) {
  int e = blockIdx.x, j = threadIdx.x;
  const float* g = geom + e * 4;
  float a = b[j] + g[1] * W[j] + g[2] * W[128 + j] + g[3] * W[256 + j];
  h[e * 128 + j] = silu_f(a);
}

// LayerNorm over 256 dims, one wave per row (8 rows / 256-thread block)
__global__ void ln_kernel(const float* __restrict__ in, float* __restrict__ out,
                          const float* __restrict__ g, const float* __restrict__ bta, int rows) {
  int row = blockIdx.x * 8 + (threadIdx.x >> 5);
  int lane = threadIdx.x & 31;
  if (row >= rows) return;
  const float* xr = in + row * 256;
  float vals[8], s = 0.0f, s2 = 0.0f;
#pragma unroll
  for (int i = 0; i < 8; ++i) {
    float v = xr[lane + i * 32];
    vals[i] = v; s += v; s2 += v * v;
  }
#pragma unroll
  for (int off = 16; off >= 1; off >>= 1) {
    s  += __shfl_xor(s,  off, 32);
    s2 += __shfl_xor(s2, off, 32);
  }
  float m = s * (1.0f / 256.0f);
  float var = s2 * (1.0f / 256.0f) - m * m;
  float inv = rsqrtf(var + 1e-5f);
#pragma unroll
  for (int i = 0; i < 8; ++i) {
    int c = lane + i * 32;
    out[row * 256 + c] = (vals[i] - m) * inv * g[c] + bta[c];
  }
}

// xacc[node] = x[node] + sum_k mask * msg[k]    (256 nodes per chunk)
__global__ void msg_reduce_kernel(const float* __restrict__ x, const float* __restrict__ msg,
                                  const float* __restrict__ maskb, float* __restrict__ xacc,
                                  int node0) {
  int ln_node = blockIdx.x, j = threadIdx.x;
  int node = node0 + ln_node;
  float acc = x[node * 256 + j];
#pragma unroll 4
  for (int k = 0; k < KNBR; ++k) {
    acc += msg[(ln_node * KNBR + k) * 256 + j] * maskb[node * KNBR + k];
  }
  xacc[node * 256 + j] = acc;
}

// attention per (b, head): K,V staged to LDS, streaming 2-pass softmax
__global__ void attn_kernel(const float* __restrict__ q, const float* __restrict__ k,
                            const float* __restrict__ v, float* __restrict__ o) {
  __shared__ float ks[NN][33];
  __shared__ float vs[NN][33];
  int bh = blockIdx.x;
  int b = bh >> 3, h = bh & 7;
  int tid = threadIdx.x;  // 192
  for (int i = tid; i < NN * 32; i += 192) {
    int n = i >> 5, d = i & 31;
    ks[n][d] = k[(b * NN + n) * 256 + h * 32 + d];
    vs[n][d] = v[(b * NN + n) * 256 + h * 32 + d];
  }
  __syncthreads();
  const float scale = 0.17677669529663687f;  // 32^-0.5
  float qv[32];
#pragma unroll
  for (int d = 0; d < 32; ++d) qv[d] = q[(b * NN + tid) * 256 + h * 32 + d];
  float mx = -3.0e38f;
  for (int j = 0; j < NN; ++j) {
    float s = 0.0f;
#pragma unroll
    for (int d = 0; d < 32; ++d) s += qv[d] * ks[j][d];
    mx = fmaxf(mx, s * scale);
  }
  float se = 0.0f, ov[32];
#pragma unroll
  for (int d = 0; d < 32; ++d) ov[d] = 0.0f;
  for (int j = 0; j < NN; ++j) {
    float s = 0.0f;
#pragma unroll
    for (int d = 0; d < 32; ++d) s += qv[d] * ks[j][d];
    float p = __expf(s * scale - mx);
    se += p;
#pragma unroll
    for (int d = 0; d < 32; ++d) ov[d] += p * vs[j][d];
  }
  float inv = 1.0f / se;
#pragma unroll
  for (int d = 0; d < 32; ++d) o[(b * NN + tid) * 256 + h * 32 + d] = ov[d] * inv;
}

// mean/max pool over N, then silu(g @ Wp + bp)
__global__ void pool_kernel(const float* __restrict__ x, const float* __restrict__ Wp,
                            const float* __restrict__ bp, float* __restrict__ out) {
  __shared__ float g[512];
  int b = blockIdx.x, j = threadIdx.x;  // 256
  float s = 0.0f, mx = -3.0e38f;
  for (int n = 0; n < NN; ++n) {
    float v = x[(b * NN + n) * 256 + j];
    s += v; mx = fmaxf(mx, v);
  }
  g[j] = s * (1.0f / (float)NN);
  g[256 + j] = mx;
  __syncthreads();
  float a = bp[j];
  for (int i = 0; i < 512; ++i) a += g[i] * Wp[i * 256 + j];
  out[b * 256 + j] = silu_f(a);
}

// ===========================================================================
// Host side
// ===========================================================================
static inline void gemm(hipStream_t s, int mode, int act,
                        const float* A, const float* W, const float* bias, const float* res,
                        float* C, int M, int N, int K, int ldC, int coff,
                        const float* gx = nullptr, const float* gef = nullptr,
                        const int* gidx = nullptr, int edge0 = 0) {
  dim3 grid(N / BN, M / BM);
  dim3 blk(256);
  if (mode == 1) {
    gemm_wmma<1, 1, false><<<grid, blk, 0, s>>>(A, W, bias, res, C, M, N, K, ldC, coff, gx, gef, gidx, edge0);
    return;
  }
  if (res) {
    gemm_wmma<0, 0, true><<<grid, blk, 0, s>>>(A, W, bias, res, C, M, N, K, ldC, coff, gx, gef, gidx, edge0);
    return;
  }
  switch (act) {
    case 0: gemm_wmma<0, 0, false><<<grid, blk, 0, s>>>(A, W, bias, res, C, M, N, K, ldC, coff, gx, gef, gidx, edge0); break;
    case 1: gemm_wmma<0, 1, false><<<grid, blk, 0, s>>>(A, W, bias, res, C, M, N, K, ldC, coff, gx, gef, gidx, edge0); break;
    default: gemm_wmma<0, 2, false><<<grid, blk, 0, s>>>(A, W, bias, res, C, M, N, K, ldC, coff, gx, gef, gidx, edge0); break;
  }
}

extern "C" void kernel_launch(void* const* d_in, const int* in_sizes, int n_in,
                              void* d_out, int out_size, void* d_ws, size_t ws_size,
                              hipStream_t stream) {
  (void)in_sizes; (void)n_in; (void)out_size; (void)ws_size;
  const float* theta = (const float*)d_in[0];
  const float* pos   = (const float*)d_in[1];
  const float* tvec  = (const float*)d_in[2];
  const int*   batch = (const int*)d_in[3];
  auto P = [&](int i) { return (const float*)d_in[i]; };

  // workspace layout (floats)
  float* ws     = (float*)d_ws;
  float* x      = ws;                         // 6144*256
  float* xacc   = x + NNODE * HH;             // 6144*256
  float* big    = xacc + NNODE * HH;          // 6144*1024 (q/k/v/o or FFN hidden)
  float* temb   = big + NNODE * 1024;         // 32*256
  float* thetas = temb + BB * HH;             // 6144*16
  float* geom   = thetas + NNODE * 16;        // ECHUNK*4
  float* hdist  = geom + ECHUNK * 4;          // ECHUNK*128
  float* hdirb  = hdist + ECHUNK * 128;       // ECHUNK*128
  float* ef     = hdirb + ECHUNK * 128;       // ECHUNK*256
  float* hidden = ef + ECHUNK * 256;          // ECHUNK*256
  float* msg    = hidden + ECHUNK * 256;      // ECHUNK*256
  float* maskb  = msg + ECHUNK * 256;         // NEDGE
  int*   idxb   = (int*)(maskb + NEDGE);      // NEDGE
  int*   flag   = idxb + NEDGE;

  // --- conditional node-type softmax ---
  zero_flag_kernel<<<1, 1, 0, stream>>>(flag);
  theta_flag_kernel<<<(NNODE * 16) / 256, 256, 0, stream>>>(theta, flag, NNODE * 16);
  theta_softmax_kernel<<<NNODE / 256, 256, 0, stream>>>(theta, thetas, flag, NNODE);

  // --- time embedding ---
  time_embed_kernel<<<1, 256, 0, stream>>>(tvec, P(6), P(7), P(8), P(9), P(10), P(11), temb);

  // --- node embedding: x = theta@Wn + bn + temb[batch] + pos-MLP ---
  node_base_kernel<<<NNODE, 256, 0, stream>>>(thetas, P(4), P(5), temb, batch, xacc);
  pos_hidden_kernel<<<NNODE, 256, 0, stream>>>(pos, P(22), P(23), hidden);
  gemm(stream, 0, 0, hidden, P(24), P(25), xacc, x, NNODE, 256, 256, 256, 0);

  // --- radius graph (KNN within cutoff) ---
  knn_kernel<<<dim3(NN, BB), 256, 0, stream>>>(pos, idxb, maskb);

  // --- GNN layers ---
  for (int L = 0; L < 3; ++L) {
    int pb = 26 + 12 * L;
    for (int c = 0; c < NEDGE / ECHUNK; ++c) {
      int edge0 = c * ECHUNK;
      int node0 = c * (ECHUNK / KNBR);
      edge_geom_kernel<<<ECHUNK / 256, 256, 0, stream>>>(pos, idxb, geom, edge0, ECHUNK);
      dist_hidden_kernel<<<ECHUNK, 128, 0, stream>>>(geom, P(12), P(13), hdist);
      dir_hidden_kernel<<<ECHUNK, 128, 0, stream>>>(geom, P(16), P(17), hdirb);
      gemm(stream, 0, 0, hdist, P(14), P(15), nullptr, ef, ECHUNK, 128, 128, 256, 0);
      gemm(stream, 0, 0, hdirb, P(18), P(19), nullptr, ef, ECHUNK, 128, 128, 256, 128);
      ln_kernel<<<ECHUNK / 8, 256, 0, stream>>>(ef, ef, P(20), P(21), ECHUNK);
      // m1 with gathered [x_i | x_j | edge_feat] input, silu
      gemm(stream, 1, 1, nullptr, P(pb + 0), P(pb + 1), nullptr, hidden,
           ECHUNK, 256, 768, 256, 0, x, ef, idxb, edge0);
      // m2 (no act)
      gemm(stream, 0, 0, hidden, P(pb + 2), P(pb + 3), nullptr, msg, ECHUNK, 256, 256, 256, 0);
      msg_reduce_kernel<<<ECHUNK / KNBR, 256, 0, stream>>>(x, msg, maskb, xacc, node0);
    }
    ln_kernel<<<NNODE / 8, 256, 0, stream>>>(xacc, x, P(pb + 4), P(pb + 5), NNODE);
    gemm(stream, 0, 1, x, P(pb + 6), P(pb + 7), nullptr, big, NNODE, 512, 256, 512, 0);
    gemm(stream, 0, 0, big, P(pb + 8), P(pb + 9), x, xacc, NNODE, 256, 512, 256, 0);
    ln_kernel<<<NNODE / 8, 256, 0, stream>>>(xacc, x, P(pb + 10), P(pb + 11), NNODE);
  }

  // --- transformer layers ---
  for (int L = 0; L < 3; ++L) {
    int pb = 62 + 16 * L;
    float* qb = big;
    float* kb = big + NNODE * HH;
    float* vb = big + 2 * NNODE * HH;
    float* ob = big + 3 * NNODE * HH;
    gemm(stream, 0, 0, x, P(pb + 0), P(pb + 1), nullptr, qb, NNODE, 256, 256, 256, 0);
    gemm(stream, 0, 0, x, P(pb + 2), P(pb + 3), nullptr, kb, NNODE, 256, 256, 256, 0);
    gemm(stream, 0, 0, x, P(pb + 4), P(pb + 5), nullptr, vb, NNODE, 256, 256, 256, 0);
    attn_kernel<<<BB * 8, NN, 0, stream>>>(qb, kb, vb, ob);
    gemm(stream, 0, 0, ob, P(pb + 6), P(pb + 7), x, xacc, NNODE, 256, 256, 256, 0);
    ln_kernel<<<NNODE / 8, 256, 0, stream>>>(xacc, x, P(pb + 8), P(pb + 9), NNODE);
    gemm(stream, 0, 2, x, P(pb + 10), P(pb + 11), nullptr, big, NNODE, 1024, 256, 1024, 0);
    gemm(stream, 0, 0, big, P(pb + 12), P(pb + 13), x, xacc, NNODE, 256, 1024, 256, 0);
    ln_kernel<<<NNODE / 8, 256, 0, stream>>>(xacc, x, P(pb + 14), P(pb + 15), NNODE);
  }

  // --- pooling + final projection ---
  pool_kernel<<<BB, 256, 0, stream>>>(x, P(110), P(111), (float*)d_out);
}
